// TemporalEncoder_13048110645693
// MI455X (gfx1250) — matrix operand order repaired and checked by reading, not compiled, not run on hardware
//
#include <hip/hip_runtime.h>
#include <hip/hip_bf16.h>
#include <stdint.h>

typedef __bf16 bf16;
typedef __attribute__((ext_vector_type(16))) __bf16 v16bf;
typedef __attribute__((ext_vector_type(8)))  __bf16 v8bf;
typedef __attribute__((ext_vector_type(8)))  float  v8f;

#define B_   64
#define T_   512
#define D_   128
#define H_   160
#define G_   480          // 3*H
#define M_   (B_*T_)      // 32768

__device__ __forceinline__ v16bf cat8(v8bf lo, v8bf hi) {
  return __builtin_shufflevector(lo, hi, 0,1,2,3,4,5,6,7,8,9,10,11,12,13,14,15);
}
__device__ __forceinline__ v8f wmma_bf16(v16bf a, v16bf b, v8f c) {
  // D = A(16x32 bf16) * B(32x16 bf16) + C(16x16 f32)
  return __builtin_amdgcn_wmma_f32_16x16x32_bf16(false, a, false, b, (short)0, c, false, false);
}
__device__ __forceinline__ float sigmoidf_(float x) { return 1.0f / (1.0f + __expf(-x)); }

// CDNA5 async global->LDS copy (16B per lane), tracked by ASYNCcnt.
// Low 32 bits of a flat LDS pointer are the LDS byte offset (aperture in high bits).
__device__ __forceinline__ void async_b128(const void* gptr, const void* lptr) {
  uint32_t lofs = (uint32_t)(uintptr_t)lptr;
  asm volatile("global_load_async_to_lds_b128 %0, %1, off"
               :: "v"(lofs), "v"(gptr) : "memory");
}
#define WAIT_ASYNC(n) asm volatile("s_wait_asynccnt " #n ::: "memory")

// ---------------------------------------------------------------- cvt f32->bf16
__global__ __launch_bounds__(256) void cvt_bf16_kernel(const float* __restrict__ in,
                                                       bf16* __restrict__ outp, int n) {
  int i = blockIdx.x * 256 + threadIdx.x;
  if (i < n) outp[i] = (bf16)in[i];
}

// ---------------------------------------------------------------- generic WMMA GEMM
// C[M,N] = act( A[M,K](bf16) * Bw[N,K]^T(bf16) + bias[N] ), fp32 out.
// 256 thr (8 waves). 128x64 C-block, 4 tiles/wave. K in 32-chunks,
// double-buffered LDS filled by global_load_async_to_lds_b128.
__global__ __launch_bounds__(256) void gemm_bf16_kernel(
    const bf16* __restrict__ A, const bf16* __restrict__ Bw,
    const float* __restrict__ bias, float* __restrict__ C,
    int N, int K, int act) {
  __shared__ bf16 As[2][128 * 32];
  __shared__ bf16 Bs[2][64 * 32];
  const int m0 = blockIdx.x * 128;
  const int n0 = blockIdx.y * 64;
  const int tid = threadIdx.x;
  const int wave = tid >> 5, lane = tid & 31;
  const int ml = lane & 15, hl = lane >> 4;
  const int mi0 = (wave >> 1) * 2;     // 0,2,4,6
  const int ni0 = (wave & 1) * 2;      // 0,2
  // fill coordinates: each thread moves 2 A rows-chunks + 1 B row-chunk (16B each)
  const int fm = tid >> 2;             // 0..63
  const int fk = (tid & 3) * 8;        // bf16 offset within 32-chunk
  const int bn = (n0 + fm < N) ? (n0 + fm) : (N - 1);   // clamp (never stored OOB)

  v8f acc00 = {}, acc01 = {}, acc10 = {}, acc11 = {};

  // prologue: stage k0 = 0 into buffer 0
  async_b128(A + (size_t)(m0 + fm) * K + fk,        &As[0][fm * 32 + fk]);
  async_b128(A + (size_t)(m0 + fm + 64) * K + fk,   &As[0][(fm + 64) * 32 + fk]);
  async_b128(Bw + (size_t)bn * K + fk,              &Bs[0][fm * 32 + fk]);

  int buf = 0;
  for (int k0 = 0; k0 < K; k0 += 32) {
    const int kn = k0 + 32;
    if (kn < K) {   // uniform branch
      async_b128(A + (size_t)(m0 + fm) * K + kn + fk,      &As[buf ^ 1][fm * 32 + fk]);
      async_b128(A + (size_t)(m0 + fm + 64) * K + kn + fk, &As[buf ^ 1][(fm + 64) * 32 + fk]);
      async_b128(Bw + (size_t)bn * K + kn + fk,            &Bs[buf ^ 1][fm * 32 + fk]);
      WAIT_ASYNC(3);    // retire the 3 older loads (current buf)
    } else {
      WAIT_ASYNC(0);
    }
    __syncthreads();
    const bf16* as = As[buf];
    const bf16* bs = Bs[buf];
    v16bf a0 = cat8(*(const v8bf*)(as + ((mi0 + 0) * 16 + ml) * 32 + hl * 16),
                    *(const v8bf*)(as + ((mi0 + 0) * 16 + ml) * 32 + hl * 16 + 8));
    v16bf a1 = cat8(*(const v8bf*)(as + ((mi0 + 1) * 16 + ml) * 32 + hl * 16),
                    *(const v8bf*)(as + ((mi0 + 1) * 16 + ml) * 32 + hl * 16 + 8));
    v16bf b0 = *(const v16bf*)(bs + ((ni0 + 0) * 16 + ml) * 32 + hl * 16);
    v16bf b1 = *(const v16bf*)(bs + ((ni0 + 1) * 16 + ml) * 32 + hl * 16);
    acc00 = wmma_bf16(a0, b0, acc00);
    acc01 = wmma_bf16(a0, b1, acc01);
    acc10 = wmma_bf16(a1, b0, acc10);
    acc11 = wmma_bf16(a1, b1, acc11);
    __syncthreads();
    buf ^= 1;
  }

  auto store_tile = [&](v8f acc, int mi, int ni) {
#pragma unroll
    for (int v = 0; v < 8; ++v) {
      const int m = m0 + mi * 16 + v + hl * 8;
      const int n = n0 + ni * 16 + ml;
      if (n < N) {
        float cv = acc[v] + bias[n];
        if (act) cv = tanhf(cv);
        C[(size_t)m * N + n] = cv;
      }
    }
  };
  store_tile(acc00, mi0, ni0);
  store_tile(acc01, mi0, ni0 + 1);
  store_tile(acc10, mi0 + 1, ni0);
  store_tile(acc11, mi0 + 1, ni0 + 1);
}

// ---------------------------------------------------------------- persistent GRU scan
// grid = (2 dirs, 4 batch-groups of 16), block = 512 (16 waves).
// LDS: Whh bf16 [480][160] | bhh | r,z,n f32 [16][160] | h bf16 [16][160] |
//      xw double-buffer f32 [2][16][480] (async-prefetched one step ahead)
#define XWBUF_OFF 191360
#define SCAN_SMEM 252800
__global__ __launch_bounds__(512) void gru_scan(
    const float* __restrict__ xw,    // [2][M_][480] precomputed x@Wih^T + bih
    const float* __restrict__ Whh,   // [2][480][160] fp32
    const float* __restrict__ bhh,   // [2][480]
    float* __restrict__ outp)        // [B_][T_][320]; this dir writes cols dir*160..
{
  extern __shared__ __align__(16) char smem[];
  bf16*  whh_s = (bf16*)smem;                    // 480*160 bf16 = 153600 B
  float* bhh_s = (float*)(smem + 153600);        // 480 f32
  float* r_s   = (float*)(smem + 155520);        // 16*160 f32
  float* z_s   = (float*)(smem + 165760);
  float* n_s   = (float*)(smem + 176000);
  bf16*  h_s   = (bf16*)(smem + 186240);         // 16*160 bf16

  const int dir = blockIdx.x;
  const int bg  = blockIdx.y;
  const int tid = threadIdx.x;
  const int wave = tid >> 5, lane = tid & 31;
  const int ml = lane & 15, hl = lane >> 4;

  const float* xwD  = xw + (size_t)dir * M_ * G_;
  const float* WhhD = Whh + (size_t)dir * G_ * H_;

  // async prefetch of one time-step's xw slice (16 rows x 480 f32 = 30 KB)
  auto stage_xw = [&](int tp, int bufi) {
    float* dst = (float*)(smem + XWBUF_OFF + bufi * 30720);
#pragma unroll
    for (int it = 0; it < 4; ++it) {
      int u = tid + it * 512;            // 2048 slots cover 1920 16B-units
      if (u > 1919) u = 1919;            // clamp: duplicate loads, uniform counts
      const int brow = u / 120, unit = u - brow * 120;
      async_b128(xwD + ((size_t)(bg * 16 + brow) * T_ + tp) * G_ + unit * 4,
                 dst + brow * 480 + unit * 4);
    }
  };

  for (int i = tid; i < G_ * H_; i += 512) whh_s[i] = (bf16)WhhD[i];
  for (int i = tid; i < G_; i += 512) bhh_s[i] = bhh[dir * G_ + i];

  float hreg[5];
#pragma unroll
  for (int k = 0; k < 5; ++k) { hreg[k] = 0.0f; h_s[tid + k * 512] = (bf16)0.0f; }

  stage_xw(dir ? (T_ - 1) : 0, 0);       // prefetch t = 0

  for (int t = 0; t < T_; ++t) {
    const int tt = dir ? (T_ - 1 - t) : t;
    WAIT_ASYNC(0);        // xw buffer for step t complete (this wave)
    __syncthreads();      // ..for all waves; also publishes h_s from step t-1
    // issue prefetch for step t+1 (overlaps with this step's compute)
    if (t + 1 < T_) stage_xw(dir ? (T_ - 2 - t) : (t + 1), (t + 1) & 1);
    const float* xwb = (const float*)(smem + XWBUF_OFF + (t & 1) * 30720);

    // ---- stage 1: gate pre-activations  g = h @ Whh^T + bhh  (16x480 via WMMA)
    for (int ni = wave; ni < 30; ni += 16) {
      v8f acc = {};
      const int arow = ml * H_;
      const int brow = (ni * 16 + ml) * H_;
#pragma unroll
      for (int kt = 0; kt < 5; ++kt) {
        const int kb = kt * 32 + hl * 16;
        v8bf alo = *(const v8bf*)(h_s + arow + kb);
        v8bf ahi = *(const v8bf*)(h_s + arow + kb + 8);
        v16bf bfrag = *(const v16bf*)(whh_s + brow + kt * 32 + hl * 16);
        acc = wmma_bf16(cat8(alo, ahi), bfrag, acc);
      }
      const int g = ni * 16 + ml;
      const int chunk = ni / 10;   // 0:r  1:z  2:n
#pragma unroll
      for (int v = 0; v < 8; ++v) {
        const int bl = v + hl * 8;               // local batch row 0..15
        float gp = acc[v] + bhh_s[g];
        if (chunk == 0) {
          gp += xwb[bl * G_ + g];
          r_s[bl * H_ + g] = sigmoidf_(gp);
        } else if (chunk == 1) {
          gp += xwb[bl * G_ + g];
          z_s[bl * H_ + (g - H_)] = sigmoidf_(gp);
        } else {
          n_s[bl * H_ + (g - 2 * H_)] = gp;      // hn pre-act (xn added later)
        }
      }
    }
    __syncthreads();
    // ---- stage 2: element update, h kept fp32 in registers
#pragma unroll
    for (int k = 0; k < 5; ++k) {
      const int idx = tid + k * 512;             // 2560 = 16*160
      const int bl = idx / H_, j = idx - bl * H_;
      const int bglob = bg * 16 + bl;
      const float xn = xwb[bl * G_ + 2 * H_ + j];
      const float nv = tanhf(xn + r_s[idx] * n_s[idx]);
      const float z  = z_s[idx];
      const float hnew = (1.0f - z) * nv + z * hreg[k];
      hreg[k] = hnew;
      h_s[idx] = (bf16)hnew;
      outp[((size_t)bglob * T_ + tt) * (2 * H_) + dir * H_ + j] = hnew;
    }
  }
}

// ---------------------------------------------------------------- motion feature
__global__ __launch_bounds__(256) void motion_kernel(const float* __restrict__ x,
                                                     float* __restrict__ motion) {
  __shared__ float sm[T_];
  __shared__ float red[256];
  const int b = blockIdx.x, tid = threadIdx.x;
  const float* xb = x + (size_t)b * T_ * D_;
  for (int t = tid; t < T_; t += 256) {
    const int ts = (t == 0) ? 1 : t;
    const float* a = xb + (size_t)ts * D_;
    const float* p = a - D_;
    float s = 0.f;
    for (int d = 0; d < D_; ++d) { float df = a[d] - p[d]; s += df * df; }
    sm[t] = sqrtf(s);
  }
  __syncthreads();
  red[tid] = sm[tid] + sm[tid + 256];
  __syncthreads();
  for (int s = 128; s > 0; s >>= 1) { if (tid < s) red[tid] += red[tid + s]; __syncthreads(); }
  const float mean = red[0] * (1.0f / T_);
  __syncthreads();
  float d0 = sm[tid] - mean, d1 = sm[tid + 256] - mean;
  red[tid] = d0 * d0 + d1 * d1;
  __syncthreads();
  for (int s = 128; s > 0; s >>= 1) { if (tid < s) red[tid] += red[tid + s]; __syncthreads(); }
  const float stdv = sqrtf(red[0] / (float)(T_ - 1));
  __syncthreads();
  const float inv = 1.0f / (stdv + 1e-6f);
  for (int t = tid; t < T_; t += 256) motion[(size_t)b * T_ + t] = (sm[t] - mean) * inv;
}

// ---------------------------------------------------------------- attn softmax over T
__global__ __launch_bounds__(512) void attn_kernel(
    const float* __restrict__ keys, const float* __restrict__ score_w,
    const float* __restrict__ score_b, const float* __restrict__ motion,
    float* __restrict__ out2) {
  __shared__ float sw[4 * H_];
  __shared__ float red[512];
  const int b = blockIdx.x, t = threadIdx.x;
  for (int i = t; i < 4 * H_; i += 512) sw[i] = score_w[i];
  __syncthreads();
  const float* kr = keys + ((size_t)b * T_ + t) * H_;
  const float mo = motion[(size_t)b * T_ + t];
  float l[4];
#pragma unroll
  for (int h = 0; h < 4; ++h) {
    float s = score_b[h];
    for (int d = 0; d < H_; ++d) s += kr[d] * sw[h * H_ + d];
    l[h] = s + mo;
  }
  float wsum = 0.f;
#pragma unroll
  for (int h = 0; h < 4; ++h) {
    red[t] = l[h]; __syncthreads();
    for (int s = 256; s > 0; s >>= 1) { if (t < s) red[t] = fmaxf(red[t], red[t + s]); __syncthreads(); }
    const float mx = red[0]; __syncthreads();
    const float e = __expf(l[h] - mx);
    red[t] = e; __syncthreads();
    for (int s = 256; s > 0; s >>= 1) { if (t < s) red[t] += red[t + s]; __syncthreads(); }
    const float sum = red[0]; __syncthreads();
    wsum += e / sum;
  }
  out2[(size_t)b * T_ + t] = wsum * 0.25f;   // == weights.mean(axis=2)
}

// ---------------------------------------------------------------- weighted pool
__global__ __launch_bounds__(320) void pool_kernel(const float* __restrict__ seq,
                                                   const float* __restrict__ out2,
                                                   float* __restrict__ pooled) {
  const int b = blockIdx.x, d = threadIdx.x;
  const float* sb = seq + (size_t)b * T_ * (2 * H_);
  const float* wb = out2 + (size_t)b * T_;
  float s = 0.f;
  for (int t = 0; t < T_; ++t) s += sb[(size_t)t * (2 * H_) + d] * wb[t];
  pooled[b * (2 * H_) + d] = s;
}

// ---------------------------------------------------------------- proj + LN + L2
__global__ __launch_bounds__(256) void head_kernel(
    const float* __restrict__ pooled, const float* __restrict__ proj_w,
    const float* __restrict__ proj_b, const float* __restrict__ ln_g,
    const float* __restrict__ ln_b, float* __restrict__ emb) {
  __shared__ float pv[2 * H_];
  __shared__ float red[256];
  const int b = blockIdx.x, d = threadIdx.x;
  for (int i = d; i < 2 * H_; i += 256) pv[i] = pooled[b * (2 * H_) + i];
  __syncthreads();
  float e = proj_b[d];
  const float* wr = proj_w + (size_t)d * (2 * H_);
  for (int k = 0; k < 2 * H_; ++k) e += pv[k] * wr[k];
  red[d] = e; __syncthreads();
  for (int s = 128; s > 0; s >>= 1) { if (d < s) red[d] += red[d + s]; __syncthreads(); }
  const float mu = red[0] * (1.0f / 256.0f); __syncthreads();
  const float c = e - mu;
  red[d] = c * c; __syncthreads();
  for (int s = 128; s > 0; s >>= 1) { if (d < s) red[d] += red[d + s]; __syncthreads(); }
  const float var = red[0] * (1.0f / 256.0f); __syncthreads();
  const float nm = c * rsqrtf(var + 1e-5f) * ln_g[d] + ln_b[d];
  red[d] = nm * nm; __syncthreads();
  for (int s = 128; s > 0; s >>= 1) { if (d < s) red[d] += red[d + s]; __syncthreads(); }
  const float nrm = fmaxf(sqrtf(red[0]), 1e-12f);
  emb[b * 256 + d] = nm / nrm;
}

// ================================================================ launch
extern "C" void kernel_launch(void* const* d_in, const int* in_sizes, int n_in,
                              void* d_out, int out_size, void* d_ws, size_t ws_size,
                              hipStream_t stream) {
  (void)in_sizes; (void)n_in; (void)out_size; (void)ws_size;
  const float* x       = (const float*)d_in[0];
  const float* Wih0    = (const float*)d_in[1];
  const float* Whh0    = (const float*)d_in[2];
  const float* bih0    = (const float*)d_in[3];
  const float* bhh0    = (const float*)d_in[4];
  const float* Wih1    = (const float*)d_in[5];
  const float* Whh1    = (const float*)d_in[6];
  const float* bih1    = (const float*)d_in[7];
  const float* bhh1    = (const float*)d_in[8];
  const float* key_w   = (const float*)d_in[9];
  const float* key_b   = (const float*)d_in[10];
  const float* score_w = (const float*)d_in[11];
  const float* score_b = (const float*)d_in[12];
  const float* proj_w  = (const float*)d_in[13];
  const float* proj_b  = (const float*)d_in[14];
  const float* ln_g    = (const float*)d_in[15];
  const float* ln_b    = (const float*)d_in[16];
  float* out = (float*)d_out;
  char* ws = (char*)d_ws;

  size_t off = 0;
  auto carve = [&](size_t bytes) { size_t o = off; off = (off + bytes + 255) & ~(size_t)255; return o; };
  float* xwp     = (float*)(ws + carve((size_t)2 * M_ * G_ * 4));     // reused layer0/layer1
  bf16*  xb      = (bf16*) (ws + carve((size_t)M_ * D_ * 2));
  bf16*  wih0b   = (bf16*) (ws + carve((size_t)2 * G_ * D_ * 2));
  bf16*  wih1b   = (bf16*) (ws + carve((size_t)2 * G_ * 2 * H_ * 2));
  bf16*  keywb   = (bf16*) (ws + carve((size_t)H_ * 2 * H_ * 2));
  float* h1      = (float*)(ws + carve((size_t)M_ * 2 * H_ * 4));
  bf16*  h1b     = (bf16*) (ws + carve((size_t)M_ * 2 * H_ * 2));
  float* seq     = (float*)(ws + carve((size_t)M_ * 2 * H_ * 4));
  bf16*  seqb    = (bf16*) (ws + carve((size_t)M_ * 2 * H_ * 2));
  float* keysp   = (float*)(ws + carve((size_t)M_ * H_ * 4));
  float* motionp = (float*)(ws + carve((size_t)B_ * T_ * 4));
  float* pooledp = (float*)(ws + carve((size_t)B_ * 2 * H_ * 4));

  (void)hipFuncSetAttribute(reinterpret_cast<const void*>(&gru_scan),
                            hipFuncAttributeMaxDynamicSharedMemorySize, SCAN_SMEM);

  // --- convert activations / weights to bf16
  cvt_bf16_kernel<<<(M_ * D_ + 255) / 256, 256, 0, stream>>>(x, xb, M_ * D_);
  cvt_bf16_kernel<<<(2 * G_ * D_ + 255) / 256, 256, 0, stream>>>(Wih0, wih0b, 2 * G_ * D_);
  cvt_bf16_kernel<<<(2 * G_ * 2 * H_ + 255) / 256, 256, 0, stream>>>(Wih1, wih1b, 2 * G_ * 2 * H_);
  cvt_bf16_kernel<<<(H_ * 2 * H_ + 255) / 256, 256, 0, stream>>>(key_w, keywb, H_ * 2 * H_);

  // --- layer 0: xW then scan
  for (int d = 0; d < 2; ++d)
    gemm_bf16_kernel<<<dim3(M_ / 128, 8), 256, 0, stream>>>(
        xb, wih0b + (size_t)d * G_ * D_, bih0 + d * G_,
        xwp + (size_t)d * M_ * G_, G_, D_, 0);
  gru_scan<<<dim3(2, 4), 512, SCAN_SMEM, stream>>>(xwp, Whh0, bhh0, h1);

  // --- layer 1: xW then scan
  cvt_bf16_kernel<<<(M_ * 2 * H_ + 255) / 256, 256, 0, stream>>>(h1, h1b, M_ * 2 * H_);
  for (int d = 0; d < 2; ++d)
    gemm_bf16_kernel<<<dim3(M_ / 128, 8), 256, 0, stream>>>(
        h1b, wih1b + (size_t)d * G_ * 2 * H_, bih1 + d * G_,
        xwp + (size_t)d * M_ * G_, G_, 2 * H_, 0);
  gru_scan<<<dim3(2, 4), 512, SCAN_SMEM, stream>>>(xwp, Whh1, bhh1, seq);

  // --- keys = tanh(seq @ key_w^T + key_b)
  cvt_bf16_kernel<<<(M_ * 2 * H_ + 255) / 256, 256, 0, stream>>>(seq, seqb, M_ * 2 * H_);
  gemm_bf16_kernel<<<dim3(M_ / 128, 3), 256, 0, stream>>>(seqb, keywb, key_b, keysp, H_, 2 * H_, 1);

  // --- attention head + outputs
  float* out2 = out + 64 * 256;   // second output region: weights.mean(axis=2), [64][512]
  motion_kernel<<<64, 256, 0, stream>>>(x, motionp);
  attn_kernel<<<64, 512, 0, stream>>>(keysp, score_w, score_b, motionp, out2);
  pool_kernel<<<64, 320, 0, stream>>>(seq, out2, pooledp);
  head_kernel<<<64, 256, 0, stream>>>(pooledp, proj_w, proj_b, ln_g, ln_b, out);
}
